// PyramidFeatures_qua_18665927869042
// MI455X (gfx1250) — matrix operand order repaired and checked by reading, not compiled, not run on hardware
//
#include <hip/hip_runtime.h>
#include <hip/hip_bf16.h>
#include <cstdint>

// ---------------------------------------------------------------------------
// Types for CDNA5 WMMA (wave32): v16bf A/B fragments, v8f accumulator.
// ---------------------------------------------------------------------------
typedef __attribute__((ext_vector_type(16))) __bf16 v16bf;
typedef __attribute__((ext_vector_type(8)))  float  v8f;
typedef int v4i_gcc __attribute__((vector_size(16)));  // matches builtin param type
using u16 = unsigned short;
using u32 = unsigned int;

union FragBF {
  v16bf bf;
  uint4 q[2];
};

__device__ __forceinline__ u16 f2bf(float f) {
  u32 u = __float_as_uint(f);
  return (u16)((u + 0x7FFFu + ((u >> 16) & 1u)) >> 16);  // round-to-nearest-even
}

// ---------------------------------------------------------------------------
// CDNA5 async global->LDS copy (ASYNCcnt path). The builtin's param 1 is
// AS1 `int __vector(4)*` (probe-confirmed by the round-2 diagnostic); LDS
// side is the AS3 equivalent. Fallback keeps compiles safe if absent.
// ---------------------------------------------------------------------------
__device__ __forceinline__ void async_b128(const u16* g, u16* l) {
#if __has_builtin(__builtin_amdgcn_global_load_async_to_lds_b128)
  typedef __attribute__((address_space(1))) v4i_gcc* gp_t;
  typedef __attribute__((address_space(3))) v4i_gcc* lp_t;
  __builtin_amdgcn_global_load_async_to_lds_b128(
      (gp_t)(unsigned long long)(const void*)g,
      (lp_t)(unsigned int)(unsigned long long)(const void*)l,
      0, 0);
#else
  *(uint4*)l = *(const uint4*)g;
#endif
}

__device__ __forceinline__ void async_wait() {
#if __has_builtin(__builtin_amdgcn_s_wait_asynccnt)
  __builtin_amdgcn_s_wait_asynccnt(0);
#elif defined(__AMDGCN__) && __has_builtin(__builtin_amdgcn_global_load_async_to_lds_b128)
  asm volatile("s_wait_asynccnt 0x0" ::: "memory");
#endif
}

// ---------------------------------------------------------------------------
// Pass 1: max(|tanh(w)|) reduction into a uint-bits slot.
// ---------------------------------------------------------------------------
__global__ __launch_bounds__(256)
void maxabs_tanh_kernel(const float* __restrict__ w, int n, u32* __restrict__ slot) {
  float m = 0.f;
  for (int i = blockIdx.x * blockDim.x + threadIdx.x; i < n; i += gridDim.x * blockDim.x)
    m = fmaxf(m, fabsf(tanhf(w[i])));
  __shared__ float red[256];
  red[threadIdx.x] = m;
  __syncthreads();
  for (int s = 128; s > 0; s >>= 1) {
    if ((int)threadIdx.x < s) red[threadIdx.x] = fmaxf(red[threadIdx.x], red[threadIdx.x + s]);
    __syncthreads();
  }
  if (threadIdx.x == 0) atomicMax(slot, __float_as_uint(red[0]));
}

// ---------------------------------------------------------------------------
// Pass 2: DoReFa quantize -> bf16, [oc][ic][tap] -> [tap][oc][ic].
// ---------------------------------------------------------------------------
__global__ __launch_bounds__(256)
void quantize_kernel(const float* __restrict__ w, u16* __restrict__ out,
                     const u32* __restrict__ slot, int OC, int K, int taps) {
  const float mx  = __uint_as_float(*slot);
  const float inv = 0.5f / mx;
  const int n = OC * K * taps;
  for (int idx = blockIdx.x * blockDim.x + threadIdx.x; idx < n; idx += gridDim.x * blockDim.x) {
    int t   = idx % taps;
    int rem = idx / taps;
    int ic  = rem % K;
    int oc  = rem / K;
    float tv = tanhf(w[idx]);
    float tt = tv * inv + 0.5f;
    float q  = rintf(tt * 255.f) * (1.f / 255.f);
    float r  = 2.f * q - 1.f;
    out[(size_t)t * OC * K + (size_t)oc * K + ic] = f2bf(r);
  }
}

// ---------------------------------------------------------------------------
// 1x1 conv as double-buffered WMMA GEMM. Block: 128 thr (4 waves),
// tile 64(M)x64(N), K-step 32, async A staging, packed b64 B staging,
// ONE barrier per K-step.
// MODE 0: +bias fp32 (P5); MODE 1: +bias+up2(prev) (P4);
// MODE 2: +bias+up2(prev), bf16 into zero-padded [66][66] (P3).
// ---------------------------------------------------------------------------
template<int MODE, int K, int W>
__global__ __launch_bounds__(128)
void gemm1x1_kernel(const u16*  __restrict__ Aw,     // [256][K] bf16
                    const float* __restrict__ X,     // [8][K][W*W] fp32
                    const float* __restrict__ bias,  // [256]
                    const float* __restrict__ prev,  // [8][256][(W/2)^2] fp32
                    float* __restrict__ outF,        // [8][256][W*W]
                    u16*   __restrict__ outBf)       // [8][256][66][66]
{
  constexpr int HW = W * W;
  constexpr int nK = K >> 5;
  const int b     = blockIdx.z;
  const int mBase = blockIdx.y * 64;
  const int pBase = blockIdx.x * 64;
  const int tid   = threadIdx.x;
  const int lane  = tid & 31;
  const int wave  = tid >> 5;

  __shared__ __align__(16) u16 lA[2][64 * 32];  // [m][k]
  __shared__ __align__(16) u16 lB[2][64 * 32];  // [n][k]

  v8f acc[4] = {};
  const float* Xb = X + (size_t)b * K * HW;

  const int am = tid >> 1, ah = tid & 1;    // A staging: 16 bf16 per thread
  const int kq = tid & 7,  seg = tid >> 3;  // B staging: k-quad x 4 n per thread

  __align__(16) float fB[4][4];

  auto stageA = [&](int kt, int buf) {
    const u16* g = Aw + (size_t)(mBase + am) * K + (kt << 5) + ah * 16;
    u16* l = &lA[buf][am * 32 + ah * 16];
    async_b128(g, l);
    async_b128(g + 8, l + 8);
  };
  auto loadB = [&](int kt) {
#pragma unroll
    for (int r = 0; r < 4; ++r)
      *(float4*)fB[r] =
          *(const float4*)(Xb + (size_t)((kt << 5) + kq * 4 + r) * HW + pBase + seg * 4);
    if (kt + 2 < nK)
      __builtin_prefetch(Xb + (size_t)((kt << 5) + 64 + kq * 4) * HW + pBase + seg * 4, 0, 0);
  };
  auto storeB = [&](int buf) {
#pragma unroll
    for (int i = 0; i < 4; ++i) {  // pack 4 k-values -> one b64 store
      u32 lo = (u32)f2bf(fB[0][i]) | ((u32)f2bf(fB[1][i]) << 16);
      u32 hi = (u32)f2bf(fB[2][i]) | ((u32)f2bf(fB[3][i]) << 16);
      *(uint2*)&lB[buf][(seg * 4 + i) * 32 + kq * 4] = make_uint2(lo, hi);
    }
  };

  stageA(0, 0);
  loadB(0);
  storeB(0);
  async_wait();
  __syncthreads();

  for (int kt = 0; kt < nK; ++kt) {
    const int buf = kt & 1;
    if (kt + 1 < nK) { stageA(kt + 1, buf ^ 1); loadB(kt + 1); }

    FragBF aF;  // 16-bit A layout: lane halves hold K {0..7,16..23}/{8..15,24..31}
    {
      const int r  = (wave << 4) | (lane & 15);
      const int kb = (lane < 16) ? 0 : 8;
      aF.q[0] = *(const uint4*)&lA[buf][r * 32 + kb];
      aF.q[1] = *(const uint4*)&lA[buf][r * 32 + kb + 16];
    }
#pragma unroll
    for (int nt = 0; nt < 4; ++nt) {
      FragBF bF;
      const int n  = nt * 16 + (lane & 15);
      const int kb = (lane < 16) ? 0 : 8;
      bF.q[0] = *(const uint4*)&lB[buf][n * 32 + kb];
      bF.q[1] = *(const uint4*)&lB[buf][n * 32 + kb + 16];
      acc[nt] = __builtin_amdgcn_wmma_f32_16x16x32_bf16(
          false, aF.bf, false, bF.bf, (short)0, acc[nt], false, false);
    }
    if (kt + 1 < nK) { storeB(buf ^ 1); async_wait(); }
    __syncthreads();
  }

  // --- epilogue: bias (+upsampled skip) (+bf16 padded store) ---
  const int col    = lane & 15;
  const int rowOff = (lane < 16) ? 0 : 8;
  float bv[8];
#pragma unroll
  for (int v = 0; v < 8; ++v) bv[v] = bias[mBase + wave * 16 + rowOff + v];

#pragma unroll
  for (int nt = 0; nt < 4; ++nt) {
    const int p = pBase + nt * 16 + col;
    const int y = p / W, x = p % W;
#pragma unroll
    for (int v = 0; v < 8; ++v) {
      const int oc = mBase + wave * 16 + rowOff + v;
      float val = acc[nt][v] + bv[v];
      if (MODE >= 1)
        val += prev[((size_t)b * 256 + oc) * (HW >> 2) + (y >> 1) * (W >> 1) + (x >> 1)];
      if (MODE == 2)
        outBf[(((size_t)b * 256 + oc) * 66 + (y + 1)) * 66 + (x + 1)] = f2bf(val);
      else
        outF[((size_t)b * 256 + oc) * HW + p] = val;
    }
  }
}

// ---------------------------------------------------------------------------
// 3x3 conv: implicit WMMA GEMM over 9 taps on zero-padded bf16 P3.
// Flattened (tap,kt) pipeline, double-buffered, async A staging.
// ---------------------------------------------------------------------------
__global__ __launch_bounds__(128)
void conv3x3_kernel(const u16* __restrict__ Aw9,   // [9][256][256] bf16
                    const u16* __restrict__ Xpad,  // [8][256][66][66] bf16
                    const float* __restrict__ bias,
                    float* __restrict__ out)       // [8][256][64][64] fp32
{
  const int b     = blockIdx.z;
  const int mBase = blockIdx.y * 64;
  const int y     = blockIdx.x;
  const int tid   = threadIdx.x;
  const int lane  = tid & 31;
  const int wave  = tid >> 5;

  __shared__ __align__(16) u16 lA[2][64 * 32];
  __shared__ __align__(16) u16 lB[2][64 * 32];

  v8f acc[4] = {};
  const u16* Xb = Xpad + (size_t)b * 256 * 66 * 66;

  const int am = tid >> 1, ah = tid & 1;     // A staging
  const int kp = tid & 15, seg = tid >> 4;   // B staging: k-pair x 8 n

  u16 hB[16];

  auto stageA = [&](int step, int buf) {
    const int tap = step >> 3, kt = step & 7;
    const u16* g = Aw9 + (size_t)tap * 256 * 256 + (size_t)(mBase + am) * 256 +
                   (kt << 5) + ah * 16;
    u16* l = &lA[buf][am * 32 + ah * 16];
    async_b128(g, l);
    async_b128(g + 8, l + 8);
  };
  auto loadB = [&](int step) {
    const int tap = step >> 3, kt = step & 7;
    const int dy = tap / 3, dx = tap % 3;
#pragma unroll
    for (int r = 0; r < 2; ++r) {
      const u16* bs = Xb + ((size_t)((kt << 5) + kp * 2 + r) * 66 + (y + dy)) * 66 +
                      seg * 8 + dx;
#pragma unroll
      for (int i = 0; i < 8; ++i) hB[r * 8 + i] = bs[i];
    }
  };
  auto storeB = [&](int buf) {
#pragma unroll
    for (int i = 0; i < 8; ++i)  // pack k-pair -> b32 store
      *(u32*)&lB[buf][(seg * 8 + i) * 32 + kp * 2] = (u32)hB[i] | ((u32)hB[8 + i] << 16);
  };

  stageA(0, 0);
  loadB(0);
  storeB(0);
  async_wait();
  __syncthreads();

  for (int step = 0; step < 72; ++step) {
    const int buf = step & 1;
    if (step + 1 < 72) { stageA(step + 1, buf ^ 1); loadB(step + 1); }

    FragBF aF;
    {
      const int r  = (wave << 4) | (lane & 15);
      const int kb = (lane < 16) ? 0 : 8;
      aF.q[0] = *(const uint4*)&lA[buf][r * 32 + kb];
      aF.q[1] = *(const uint4*)&lA[buf][r * 32 + kb + 16];
    }
#pragma unroll
    for (int nt = 0; nt < 4; ++nt) {
      FragBF bF;
      const int n  = nt * 16 + (lane & 15);
      const int kb = (lane < 16) ? 0 : 8;
      bF.q[0] = *(const uint4*)&lB[buf][n * 32 + kb];
      bF.q[1] = *(const uint4*)&lB[buf][n * 32 + kb + 16];
      acc[nt] = __builtin_amdgcn_wmma_f32_16x16x32_bf16(
          false, aF.bf, false, bF.bf, (short)0, acc[nt], false, false);
    }
    if (step + 1 < 72) { storeB(buf ^ 1); async_wait(); }
    __syncthreads();
  }

  const int col    = lane & 15;
  const int rowOff = (lane < 16) ? 0 : 8;
  float bv[8];
#pragma unroll
  for (int v = 0; v < 8; ++v) bv[v] = bias[mBase + wave * 16 + rowOff + v];

#pragma unroll
  for (int nt = 0; nt < 4; ++nt) {
    const int x = nt * 16 + col;
#pragma unroll
    for (int v = 0; v < 8; ++v) {
      const int oc = mBase + wave * 16 + rowOff + v;
      out[(((size_t)b * 256 + oc) * 64 + y) * 64 + x] = acc[nt][v] + bv[v];
    }
  }
}

// ---------------------------------------------------------------------------
// Host launcher. Inputs: C3,C4,C5,w5,b5,w4,b4,w3,b3,w3_2,b3_2.
// ---------------------------------------------------------------------------
extern "C" void kernel_launch(void* const* d_in, const int* in_sizes, int n_in,
                              void* d_out, int out_size, void* d_ws, size_t ws_size,
                              hipStream_t stream) {
  (void)in_sizes; (void)n_in; (void)out_size; (void)ws_size;
  const float* C3  = (const float*)d_in[0];
  const float* C4  = (const float*)d_in[1];
  const float* C5  = (const float*)d_in[2];
  const float* w5  = (const float*)d_in[3];
  const float* b5  = (const float*)d_in[4];
  const float* w4  = (const float*)d_in[5];
  const float* b4  = (const float*)d_in[6];
  const float* w3  = (const float*)d_in[7];
  const float* b3  = (const float*)d_in[8];
  const float* w32 = (const float*)d_in[9];
  const float* b32 = (const float*)d_in[10];
  float* out = (float*)d_out;

  char* ws = (char*)d_ws;
  size_t off = 0;
  auto alloc = [&](size_t bytes) -> void* {
    void* p = ws + off;
    off = (off + bytes + 255) & ~(size_t)255;
    return p;
  };
  u32* slots = (u32*)alloc(256);
  u16* wq5   = (u16*)alloc((size_t)256 * 2048 * 2);
  u16* wq4   = (u16*)alloc((size_t)256 * 1024 * 2);
  u16* wq3   = (u16*)alloc((size_t)256 * 512 * 2);
  u16* wq9   = (u16*)alloc((size_t)9 * 256 * 256 * 2);
  float* P5  = (float*)alloc((size_t)8 * 256 * 256 * 4);
  float* P4  = (float*)alloc((size_t)8 * 256 * 1024 * 4);
  u16*  P3p  = (u16*)alloc((size_t)8 * 256 * 66 * 66 * 2);

  (void)hipMemsetAsync(slots, 0, 256, stream);
  (void)hipMemsetAsync(P3p, 0, (size_t)8 * 256 * 66 * 66 * 2, stream);

  maxabs_tanh_kernel<<<128, 256, 0, stream>>>(w5, 256 * 2048, slots + 0);
  maxabs_tanh_kernel<<<128, 256, 0, stream>>>(w4, 256 * 1024, slots + 1);
  maxabs_tanh_kernel<<<64, 256, 0, stream>>>(w3, 256 * 512, slots + 2);
  maxabs_tanh_kernel<<<128, 256, 0, stream>>>(w32, 256 * 256 * 9, slots + 3);

  quantize_kernel<<<512, 256, 0, stream>>>(w5, wq5, slots + 0, 256, 2048, 1);
  quantize_kernel<<<256, 256, 0, stream>>>(w4, wq4, slots + 1, 256, 1024, 1);
  quantize_kernel<<<128, 256, 0, stream>>>(w3, wq3, slots + 2, 256, 512, 1);
  quantize_kernel<<<512, 256, 0, stream>>>(w32, wq9, slots + 3, 256, 256, 9);

  gemm1x1_kernel<0, 2048, 16><<<dim3(4, 4, 8), 128, 0, stream>>>(
      wq5, C5, b5, nullptr, P5, nullptr);
  gemm1x1_kernel<1, 1024, 32><<<dim3(16, 4, 8), 128, 0, stream>>>(
      wq4, C4, b4, P5, P4, nullptr);
  gemm1x1_kernel<2, 512, 64><<<dim3(64, 4, 8), 128, 0, stream>>>(
      wq3, C3, b3, P4, nullptr, P3p);
  conv3x3_kernel<<<dim3(64, 4, 8), 128, 0, stream>>>(wq9, P3p, b32, out);
}
